// Attention_55233279426826
// MI455X (gfx1250) — matrix operand order repaired and checked by reading, not compiled, not run on hardware
//
#include <hip/hip_runtime.h>
#include <hip/hip_bf16.h>

typedef __attribute__((ext_vector_type(16))) __bf16 v16bf;
typedef __attribute__((ext_vector_type(8)))  __bf16 v8bf;
typedef __attribute__((ext_vector_type(8)))  float  v8f;

#define NPATCH 64
#define NWIN   32
#define HEADS  12
#define DH     64
#define DIM    768
#define BATCH  2
#define NTOK   2048          // NPATCH*NWIN
#define BN     4096          // BATCH*NTOK
#define QKVN   2304          // 3*DIM

// ---------------------------------------------------------------- WMMA helpers

__device__ __forceinline__ v8f wmma_bf16(v16bf a, v16bf b, v8f c) {
  // D = A(16x32) * B(32x16) + C, f32 accumulate
  return __builtin_amdgcn_wmma_f32_16x16x32_bf16(
      false, a, false, b, (short)0, c, false, false);
}

// A fragment (16x32) from row-major src[m][k], leading dim ld (elements).
// ISA layout: lane holds row (l&15); elems 0..7 -> K=kb..kb+7, elems 8..15 ->
// K=kb+16..kb+23, where kb = k0 + 8*(lane>=16).
__device__ __forceinline__ v16bf load_A16x32(const __bf16* src, int ld, int m0, int k0) {
  int lane = threadIdx.x & 31;
  const __bf16* p = src + (m0 + (lane & 15)) * ld + k0 + ((lane & 16) >> 1);
  v8bf lo = *(const v8bf*)p;
  v8bf hi = *(const v8bf*)(p + 16);
  v16bf a;
#pragma unroll
  for (int t = 0; t < 8; ++t) { a[t] = lo[t]; a[8 + t] = hi[t]; }
  return a;
}

// B fragment (32x16) where memory holds B^T row-major: src[n][k], leading dim ld.
// ISA layout: lane holds column (l&15); elems 0..7 -> K=kb..kb+7, elems 8..15 ->
// K=kb+16..kb+23, where kb = k0 + 8*(lane>=16).
__device__ __forceinline__ v16bf load_B_from_T(const __bf16* src, int ld, int n0, int k0) {
  int lane = threadIdx.x & 31;
  const __bf16* p = src + (n0 + (lane & 15)) * ld + k0 + ((lane & 16) >> 1);
  v8bf lo = *(const v8bf*)p;
  v8bf hi = *(const v8bf*)(p + 16);
  v16bf b;
#pragma unroll
  for (int t = 0; t < 8; ++t) { b[t] = lo[t]; b[8 + t] = hi[t]; }
  return b;
}

// Shared K-pipelined 32x64 GEMM core: acc += A[m0..m0+31][:] * W^T[n0..n0+63][:]
// Ping-pong double buffering, manually unrolled by 2 so no register copies.
__device__ __forceinline__ void gemm_kloop(const __bf16* __restrict__ A,
                                           const __bf16* __restrict__ Wt,
                                           int m0, int n0, v8f acc[2][4]) {
  v16bf a0[2], a1[2], b0v[4], b1v[4];
  a0[0] = load_A16x32(A, DIM, m0, 0);
  a0[1] = load_A16x32(A, DIM, m0 + 16, 0);
#pragma unroll
  for (int c = 0; c < 4; ++c) b0v[c] = load_B_from_T(Wt, DIM, n0 + 16 * c, 0);

  for (int kt = 0; kt < DIM / 32; kt += 2) {
    int kn1 = (kt + 1) * 32;                                  // always < DIM
    a1[0] = load_A16x32(A, DIM, m0, kn1);
    a1[1] = load_A16x32(A, DIM, m0 + 16, kn1);
#pragma unroll
    for (int c = 0; c < 4; ++c) b1v[c] = load_B_from_T(Wt, DIM, n0 + 16 * c, kn1);
#pragma unroll
    for (int c = 0; c < 4; ++c) {
      acc[0][c] = wmma_bf16(a0[0], b0v[c], acc[0][c]);
      acc[1][c] = wmma_bf16(a0[1], b0v[c], acc[1][c]);
    }
    int kn2 = (kt + 2 < DIM / 32) ? (kt + 2) * 32 : 0;        // wraparound dummy
    a0[0] = load_A16x32(A, DIM, m0, kn2);
    a0[1] = load_A16x32(A, DIM, m0 + 16, kn2);
#pragma unroll
    for (int c = 0; c < 4; ++c) b0v[c] = load_B_from_T(Wt, DIM, n0 + 16 * c, kn2);
#pragma unroll
    for (int c = 0; c < 4; ++c) {
      acc[0][c] = wmma_bf16(a1[0], b1v[c], acc[0][c]);
      acc[1][c] = wmma_bf16(a1[1], b1v[c], acc[1][c]);
    }
  }
}

// ---------------------------------------------------------------- LayerNorm

__global__ __launch_bounds__(256) void ln_kernel(const float* __restrict__ x,
                                                 const float* __restrict__ gamma,
                                                 const float* __restrict__ beta,
                                                 __bf16* __restrict__ xn) {
  int row = blockIdx.x;            // 0..BN-1
  int t   = threadIdx.x;           // 0..255
  const float* xr = x + (long)row * DIM;
  float v0 = xr[t], v1 = xr[t + 256], v2 = xr[t + 512];
  float s  = v0 + v1 + v2;
  float s2 = v0 * v0 + v1 * v1 + v2 * v2;
#pragma unroll
  for (int m = 16; m >= 1; m >>= 1) {
    s  += __shfl_xor(s,  m, 32);
    s2 += __shfl_xor(s2, m, 32);
  }
  __shared__ float rs[8], rs2[8];
  int wid = t >> 5, lane = t & 31;
  if (lane == 0) { rs[wid] = s; rs2[wid] = s2; }
  __syncthreads();
  if (t == 0) {
    float a = 0.f, b = 0.f;
#pragma unroll
    for (int i = 0; i < 8; ++i) { a += rs[i]; b += rs2[i]; }
    rs[0] = a; rs2[0] = b;
  }
  __syncthreads();
  float mu  = rs[0] * (1.0f / DIM);
  float var = rs2[0] * (1.0f / DIM) - mu * mu;
  float inv = rsqrtf(var + 1e-5f);
  __bf16* xo = xn + (long)row * DIM;
  xo[t]       = (__bf16)((v0 - mu) * inv * gamma[t]       + beta[t]);
  xo[t + 256] = (__bf16)((v1 - mu) * inv * gamma[t + 256] + beta[t + 256]);
  xo[t + 512] = (__bf16)((v2 - mu) * inv * gamma[t + 512] + beta[t + 512]);
}

// -------------------------------------------- weight convert (fp32 -> bf16, transpose)

__global__ __launch_bounds__(256) void cvt_w_kernel(const float* __restrict__ wqkv,
                                                    const float* __restrict__ wout,
                                                    __bf16* __restrict__ wqkv_t,
                                                    __bf16* __restrict__ wout_t) {
  int idx = blockIdx.x * blockDim.x + threadIdx.x;
  if (idx < DIM * QKVN) {                // wqkv [DIM][QKVN] -> wqkv_t [QKVN][DIM]
    int k = idx / QKVN, n = idx % QKVN;
    wqkv_t[(long)n * DIM + k] = (__bf16)wqkv[idx];
  }
  if (idx < DIM * DIM) {                 // wout [DIM][DIM] -> wout_t [DIM][DIM]
    int k = idx / DIM, n = idx % DIM;
    wout_t[(long)n * DIM + k] = (__bf16)wout[idx];
  }
}

// ---------------------------------------------------------------- QKV GEMM
// C[BN x QKVN] = xn[BN x DIM] * wqkv ; one wave per 32x64 strip, K-pipelined.
// q,k stored per-head [b][h][n][DH]; v stored TRANSPOSED [b][h][DH][n] so the
// attention P@V B-fragments are contiguous b128 loads.

__global__ __launch_bounds__(256) void qkv_gemm_kernel(const __bf16* __restrict__ xn,
                                                       const __bf16* __restrict__ wt,
                                                       __bf16* __restrict__ qb,
                                                       __bf16* __restrict__ kb,
                                                       __bf16* __restrict__ vt) {
  int gwave = blockIdx.x * (blockDim.x >> 5) + (threadIdx.x >> 5); // 0..4607
  int mt = gwave / 36, ng = gwave % 36;      // 128 m-tiles(32) x 36 n-groups(64)
  int m0 = mt * 32, n0 = ng * 64;

  v8f acc[2][4];
#pragma unroll
  for (int mi = 0; mi < 2; ++mi)
#pragma unroll
    for (int c = 0; c < 4; ++c)
#pragma unroll
      for (int i = 0; i < 8; ++i) acc[mi][c][i] = 0.f;

  gemm_kloop(xn, wt, m0, n0, acc);

  int lane  = threadIdx.x & 31;
  int l15   = lane & 15;
  int half8 = (lane & 16) >> 1;
  int g     = n0 >> 6;                 // 64-col block id
  int which = g / HEADS;               // 0=q,1=k,2=v (uniform for whole strip)
  int h     = g % HEADS;

  if (which == 2) {                    // V: transposed [b][h][DH][NTOK]
#pragma unroll
    for (int mi = 0; mi < 2; ++mi)
#pragma unroll
      for (int c = 0; c < 4; ++c)
#pragma unroll
        for (int vi = 0; vi < 8; ++vi) {
          int m  = m0 + 16 * mi + vi + half8;
          int b_ = m >> 11;
          int n  = m & (NTOK - 1);
          int d  = 16 * c + l15;
          vt[(((long)(b_ * HEADS + h)) * DH + d) * NTOK + n] = (__bf16)acc[mi][c][vi];
        }
  } else {                             // Q or K: [b][h][NTOK][DH]
    __bf16* dst = (which == 0) ? qb : kb;
#pragma unroll
    for (int mi = 0; mi < 2; ++mi)
#pragma unroll
      for (int c = 0; c < 4; ++c)
#pragma unroll
        for (int vi = 0; vi < 8; ++vi) {
          int m  = m0 + 16 * mi + vi + half8;
          int b_ = m >> 11;
          int n  = m & (NTOK - 1);
          int d  = 16 * c + l15;
          dst[(((long)(b_ * HEADS + h)) * NTOK + n) * DH + d] = (__bf16)acc[mi][c][vi];
        }
  }
}

// ---------------------------------------------------------------- Attention
// One WG (4 waves) per (b*h, query-window). Wave w owns rows w*16..w*16+15.
// Flash-style online softmax over key windows j <= iw (block-causal mask).

__global__ __launch_bounds__(128) void attn_kernel(const __bf16* __restrict__ q,
                                                   const __bf16* __restrict__ k,
                                                   const __bf16* __restrict__ vt,
                                                   __bf16* __restrict__ oh) {
  int bid = blockIdx.x;                 // 0..BATCH*HEADS*NWIN-1
  int iw  = bid & (NWIN - 1);
  int bh  = bid >> 5;                   // b*HEADS + h
  int wv  = threadIdx.x >> 5;
  int lane = threadIdx.x & 31;
  int l15 = lane & 15;
  int half8 = (lane & 16) >> 1;

  const __bf16* Q  = q  + (long)bh * NTOK * DH;
  const __bf16* K  = k  + (long)bh * NTOK * DH;
  const __bf16* Vt = vt + (long)bh * DH * NTOK;   // [DH][NTOK]

  __shared__ __align__(16) __bf16 Plds[4][16][DH];

  int qrow0 = iw * NPATCH + wv * 16;
  v16bf qa0 = load_A16x32(Q, DH, qrow0, 0);
  v16bf qa1 = load_A16x32(Q, DH, qrow0, 32);

  v8f o[4];
  float mrow[8], lsum[8];
#pragma unroll
  for (int c = 0; c < 4; ++c)
#pragma unroll
    for (int i = 0; i < 8; ++i) o[c][i] = 0.f;
#pragma unroll
  for (int i = 0; i < 8; ++i) { mrow[i] = -3.0e38f; lsum[i] = 0.f; }

  const float scale = 0.125f;           // DIM_HEAD^-0.5

  for (int j = 0; j <= iw; ++j) {
    int krow0 = j * NPATCH;
    if (j < iw) {                       // prefetch next window (global_prefetch_b8)
      __builtin_prefetch(K + (long)(krow0 + NPATCH + l15) * DH, 0, 1);
      __builtin_prefetch(Vt + (long)l15 * NTOK + krow0 + NPATCH, 0, 1);
    }
    // ---- S = Q K^T (16x64 band, 4 col tiles)
    v8f s[4];
#pragma unroll
    for (int c = 0; c < 4; ++c) {
      v16bf b0 = load_B_from_T(K, DH, krow0 + 16 * c, 0);
      v16bf b1 = load_B_from_T(K, DH, krow0 + 16 * c, 32);
      v8f z;
#pragma unroll
      for (int i = 0; i < 8; ++i) z[i] = 0.f;
      z = wmma_bf16(qa0, b0, z);
      z = wmma_bf16(qa1, b1, z);
      s[c] = z;
    }
    // ---- V fragments in flight while softmax VALU work runs
    v16bf vbf[2][4];
#pragma unroll
    for (int c = 0; c < 4; ++c) {
      vbf[0][c] = load_B_from_T(Vt, NTOK, 16 * c, krow0);
      vbf[1][c] = load_B_from_T(Vt, NTOK, 16 * c, krow0 + 32);
    }
    // ---- online softmax stats (per row = vgpr idx within lane half)
    float rmax[8];
#pragma unroll
    for (int i = 0; i < 8; ++i)
      rmax[i] = fmaxf(fmaxf(s[0][i], s[1][i]), fmaxf(s[2][i], s[3][i]));
#pragma unroll
    for (int mk = 8; mk >= 1; mk >>= 1)
#pragma unroll
      for (int i = 0; i < 8; ++i)
        rmax[i] = fmaxf(rmax[i], __shfl_xor(rmax[i], mk, 32));

    float fac[8];
#pragma unroll
    for (int i = 0; i < 8; ++i) {
      float mn = fmaxf(mrow[i], rmax[i] * scale);
      fac[i] = __expf(mrow[i] - mn);    // underflows to 0 for initial -inf-ish
      mrow[i] = mn;
    }
    // ---- P = exp(scale*S - m); stage to LDS (private per-wave strip)
#pragma unroll
    for (int c = 0; c < 4; ++c)
#pragma unroll
      for (int i = 0; i < 8; ++i) {
        float p = __expf(s[c][i] * scale - mrow[i]);
        s[c][i] = p;
        Plds[wv][i + half8][16 * c + l15] = (__bf16)p;
      }
#pragma unroll
    for (int i = 0; i < 8; ++i) {
      float t = s[0][i] + s[1][i] + s[2][i] + s[3][i];
#pragma unroll
      for (int mk = 8; mk >= 1; mk >>= 1) t += __shfl_xor(t, mk, 32);
      lsum[i] = lsum[i] * fac[i] + t;
    }
#pragma unroll
    for (int c = 0; c < 4; ++c)
#pragma unroll
      for (int i = 0; i < 8; ++i) o[c][i] *= fac[i];

    __syncthreads();
    // ---- reload P in A-fragment layout, O += P @ V
    v16bf pa0 = load_A16x32(&Plds[wv][0][0], DH, 0, 0);
    v16bf pa1 = load_A16x32(&Plds[wv][0][0], DH, 0, 32);
#pragma unroll
    for (int c = 0; c < 4; ++c) {
      o[c] = wmma_bf16(pa0, vbf[0][c], o[c]);
      o[c] = wmma_bf16(pa1, vbf[1][c], o[c]);
    }
    __syncthreads();
  }

  // ---- normalize + write out_heads [b][n][h*DH + d] bf16
  int b_ = bh / HEADS, h = bh % HEADS;
#pragma unroll
  for (int c = 0; c < 4; ++c)
#pragma unroll
    for (int i = 0; i < 8; ++i) {
      int row = qrow0 + i + half8;
      int d   = 16 * c + l15;
      float val = o[c][i] / lsum[i];
      oh[((long)(b_ * NTOK) + row) * DIM + h * DH + d] = (__bf16)val;
    }
}

// ---------------------------------------------------------------- output GEMM
// out[BN x DIM] = oh[BN x DIM] * w_out + b_out, fp32 output. 32x64 per wave,
// K-pipelined like the QKV GEMM.

__global__ __launch_bounds__(256) void out_gemm_kernel(const __bf16* __restrict__ oh,
                                                       const __bf16* __restrict__ wt,
                                                       const float* __restrict__ bias,
                                                       float* __restrict__ out) {
  int gwave = blockIdx.x * (blockDim.x >> 5) + (threadIdx.x >> 5); // 0..1535
  int mt = gwave / 12, ng = gwave % 12;       // 128 m-tiles(32) x 12 n-groups(64)
  int m0 = mt * 32, n0 = ng * 64;

  v8f acc[2][4];
#pragma unroll
  for (int mi = 0; mi < 2; ++mi)
#pragma unroll
    for (int c = 0; c < 4; ++c)
#pragma unroll
      for (int i = 0; i < 8; ++i) acc[mi][c][i] = 0.f;

  gemm_kloop(oh, wt, m0, n0, acc);

  int lane = threadIdx.x & 31;
  int l15 = lane & 15;
  int half8 = (lane & 16) >> 1;
#pragma unroll
  for (int mi = 0; mi < 2; ++mi)
#pragma unroll
    for (int c = 0; c < 4; ++c)
#pragma unroll
      for (int vi = 0; vi < 8; ++vi) {
        int m = m0 + 16 * mi + vi + half8;
        int n = n0 + 16 * c + l15;
        out[(long)m * DIM + n] = acc[mi][c][vi] + bias[n];
      }
}

// ---------------------------------------------------------------- launch

extern "C" void kernel_launch(void* const* d_in, const int* in_sizes, int n_in,
                              void* d_out, int out_size, void* d_ws, size_t ws_size,
                              hipStream_t stream) {
  const float* x     = (const float*)d_in[0];
  const float* gamma = (const float*)d_in[1];
  const float* beta  = (const float*)d_in[2];
  const float* w_qkv = (const float*)d_in[3];
  const float* w_out = (const float*)d_in[4];
  const float* b_out = (const float*)d_in[5];
  float* out = (float*)d_out;

  char* ws = (char*)d_ws;
  size_t off = 0;
  __bf16* xn     = (__bf16*)(ws + off); off += (size_t)BN * DIM * 2;      // 6.29 MB
  __bf16* wqkv_t = (__bf16*)(ws + off); off += (size_t)QKVN * DIM * 2;    // 3.54 MB
  __bf16* wout_t = (__bf16*)(ws + off); off += (size_t)DIM * DIM * 2;     // 1.18 MB
  __bf16* qb     = (__bf16*)(ws + off); off += (size_t)BATCH * HEADS * NTOK * DH * 2;
  __bf16* kb     = (__bf16*)(ws + off); off += (size_t)BATCH * HEADS * NTOK * DH * 2;
  __bf16* vt     = (__bf16*)(ws + off); off += (size_t)BATCH * HEADS * DH * NTOK * 2;
  __bf16* oh     = (__bf16*)(ws + off); off += (size_t)BN * DIM * 2;

  ln_kernel<<<BN, 256, 0, stream>>>(x, gamma, beta, xn);

  int cvt_total = DIM * QKVN;
  cvt_w_kernel<<<(cvt_total + 255) / 256, 256, 0, stream>>>(w_qkv, w_out, wqkv_t, wout_t);

  // 128 m-tiles * 36 n-groups = 4608 waves, 8 waves/block
  qkv_gemm_kernel<<<4608 / 8, 256, 0, stream>>>(xn, wqkv_t, qb, kb, vt);

  attn_kernel<<<BATCH * HEADS * NWIN, 128, 0, stream>>>(qb, kb, vt, oh);

  // 128 m-tiles * 12 n-groups = 1536 waves, 8 waves/block
  out_gemm_kernel<<<1536 / 8, 256, 0, stream>>>(oh, wout_t, b_out, out);
}
